// SwinTransformerBlock_34591666602375
// MI455X (gfx1250) — compile-verified, matrix operand order checked by reading, not run on hardware
//
#include <hip/hip_runtime.h>
#include <math.h>

// ---------------- types / helpers ----------------
typedef __attribute__((ext_vector_type(16))) __bf16    v16bf;
typedef __attribute__((ext_vector_type(8)))  float     v8f;
typedef __attribute__((ext_vector_type(4)))  unsigned  u32x4;
typedef __attribute__((ext_vector_type(4)))  int       i32x4;
typedef __attribute__((ext_vector_type(4)))  float     f32x4;

union FragU { v16bf v; unsigned u[8]; u32x4 q[2]; };

__device__ __forceinline__ unsigned short f2bf(float f) {
    unsigned u = __builtin_bit_cast(unsigned, f);
    u += 0x7FFFu + ((u >> 16) & 1u);          // round-to-nearest-even
    return (unsigned short)(u >> 16);
}

// single-instruction xor-lane exchange (group-of-32 ds_swizzle, and_mask=0x1f)
template<int XMASK>
__device__ __forceinline__ float swz_xor(float x) {
    return __builtin_bit_cast(float,
        __builtin_amdgcn_ds_swizzle(__builtin_bit_cast(int, x), (XMASK << 10) | 0x1F));
}
__device__ __forceinline__ float wave_sum(float x) {
    x += swz_xor<16>(x); x += swz_xor<8>(x); x += swz_xor<4>(x);
    x += swz_xor<2>(x);  x += swz_xor<1>(x);
    return x;
}

// ---- async global->LDS staging (CDNA5 ASYNCcnt path), with sync fallback ----
#if defined(__has_builtin)
#if __has_builtin(__builtin_amdgcn_global_load_async_to_lds_b128)
#define HAVE_ASYNC_LDS 1
#endif
#endif

__device__ __forceinline__ void copy16_g2l(const unsigned short* gp, unsigned short* lp) {
#if defined(HAVE_ASYNC_LDS)
    __builtin_amdgcn_global_load_async_to_lds_b128(
        (__attribute__((address_space(1))) i32x4*)gp,
        (__attribute__((address_space(3))) i32x4*)lp, 0, 0);
#else
    *(u32x4*)lp = *(const u32x4*)gp;
#endif
}
__device__ __forceinline__ void wait_async_lds() {
#if defined(HAVE_ASYNC_LDS)
#if __has_builtin(__builtin_amdgcn_s_wait_asynccnt)
    __builtin_amdgcn_s_wait_asynccnt(0);
#else
    asm volatile("s_wait_asynccnt 0x0" ::: "memory");
#endif
#endif
}

#define DIMC   256
#define HEADS  8
#define DHEAD  32
#define NTOK   64
#define NWIN   64
#define NROWS  65536                    // B*H*W tokens
#define SCALE_Q 0.17677669529663687f    // 32^-0.5

// ---------------- kernel: f32 -> bf16 cast ----------------
__global__ __launch_bounds__(256)
void cast_f32_bf16(const float* __restrict__ src, unsigned short* __restrict__ dst, int n) {
    int i = blockIdx.x * 256 + threadIdx.x;
    if (i < n) dst[i] = f2bf(src[i]);
}

// ---------------- kernel: expand rpb -> biasT[h][q][k] (8x64x64 f32) ----------------
__global__ __launch_bounds__(256)
void bias_prep(const float* __restrict__ rpb, float* __restrict__ biasT) {
    int idx = blockIdx.x * 256 + threadIdx.x;          // 32768
    int h = idx >> 12, q = (idx >> 6) & 63, k = idx & 63;
    int rel = ((q >> 3) - (k >> 3) + 7) * 15 + ((q & 7) - (k & 7) + 7);
    biasT[idx] = rpb[rel * HEADS + h];
}

// ---------------- kernel: LN1 + shift + window partition (x0 & x1) ----------------
__global__ __launch_bounds__(256)
void ln_shift_window(const float* __restrict__ x0, const float* __restrict__ x1,
                     const float* __restrict__ g0, const float* __restrict__ b0,
                     const float* __restrict__ g1, const float* __restrict__ b1,
                     unsigned short* __restrict__ x0w, unsigned short* __restrict__ x1w) {
    int p = blockIdx.x;                // b*4096 + i*64 + j
    int c = threadIdx.x;               // channel
    int b = p >> 12, ij = p & 4095, i = ij >> 6, j = ij & 63;
    float v0 = x0[(size_t)p * DIMC + c];
    float v1 = x1[(size_t)p * DIMC + c];
    float s0 = wave_sum(v0), s1 = wave_sum(v1);
    float q0 = wave_sum(v0 * v0), q1 = wave_sum(v1 * v1);
    __shared__ float red[4][8];
    int wave = c >> 5, lane = c & 31;
    if (lane == 0) { red[0][wave] = s0; red[1][wave] = s1; red[2][wave] = q0; red[3][wave] = q1; }
    __syncthreads();
    s0 = s1 = q0 = q1 = 0.f;
    #pragma unroll
    for (int w = 0; w < 8; ++w) { s0 += red[0][w]; s1 += red[1][w]; q0 += red[2][w]; q1 += red[3][w]; }
    float m0 = s0 * (1.f/256.f), m1 = s1 * (1.f/256.f);
    float r0 = rsqrtf(q0 * (1.f/256.f) - m0 * m0 + 1e-5f);
    float r1 = rsqrtf(q1 * (1.f/256.f) - m1 * m1 + 1e-5f);
    float y0 = (v0 - m0) * r0 * g0[c] + b0[c];
    float y1 = (v1 - m1) * r1 * g1[c] + b1[c];
    int is = (i - 4) & 63, js = (j - 4) & 63;           // roll(-4)
    int g = b * NWIN + (is >> 3) * 8 + (js >> 3);
    int n = (is & 7) * 8 + (js & 7);
    size_t o = ((size_t)g * NTOK + n) * DIMC + c;
    x0w[o] = f2bf(y0);
    x1w[o] = f2bf(y1);
}

// ---------------- kernel: LN2 ----------------
__global__ __launch_bounds__(256)
void ln2_kernel(const float* __restrict__ x, const float* __restrict__ g,
                const float* __restrict__ b, unsigned short* __restrict__ out) {
    int p = blockIdx.x, c = threadIdx.x;
    float v = x[(size_t)p * DIMC + c];
    float s = wave_sum(v), q = wave_sum(v * v);
    __shared__ float red[2][8];
    int wave = c >> 5, lane = c & 31;
    if (lane == 0) { red[0][wave] = s; red[1][wave] = q; }
    __syncthreads();
    s = q = 0.f;
    #pragma unroll
    for (int w = 0; w < 8; ++w) { s += red[0][w]; q += red[1][w]; }
    float m = s * (1.f/256.f);
    float r = rsqrtf(q * (1.f/256.f) - m * m + 1e-5f);
    out[(size_t)p * DIMC + c] = f2bf((v - m) * r * g[c] + b[c]);
}

// ---------------- generic bf16 WMMA GEMM: out = A[M,K] @ W[N,K]^T (+epilogue) ----
// W tile (128 cols x 32 k) is staged into double-buffered LDS with async
// global->LDS loads (ASYNCcnt) pipelined one k-step ahead of the WMMAs.
template<int K, int EPI>
__global__ __launch_bounds__(256)
void gemm_bf16_ws(const unsigned short* __restrict__ A,
                  const unsigned short* __restrict__ Wb,
                  const float* __restrict__ bias,
                  const float* __restrict__ aux,
                  void* __restrict__ outp) {
    __shared__ unsigned short Wt[2][128 * 32];          // 16 KB double buffer
    int tid = threadIdx.x;
    int lane = tid & 31, wave = tid >> 5;
    int mbase = blockIdx.x * 64 + (wave & 3) * 16;
    int nblk  = blockIdx.y * 128;
    int nwave = (wave >> 2) * 64;                       // 0 or 64 within the 128-col tile
    int am = mbase + (lane & 15);
    int hsel = (lane >> 4) << 3;                        // +8 elements for upper half-wave

    // 512 16-byte chunks per tile, 2 per thread: chunk = col*4 + part
    int c0 = tid,  col0 = c0 >> 2, part0 = (c0 & 3) << 3;
    int c1 = tid + 256, col1 = c1 >> 2, part1 = (c1 & 3) << 3;

    // stage first tile
    copy16_g2l(Wb + (size_t)(nblk + col0) * K + part0, &Wt[0][col0 * 32 + part0]);
    copy16_g2l(Wb + (size_t)(nblk + col1) * K + part1, &Wt[0][col1 * 32 + part1]);

    v8f acc[4] = {};
    for (int kk = 0, it = 0; kk < K; kk += 32, ++it) {
        int buf = it & 1;
        wait_async_lds();
        __syncthreads();                                // tile[buf] ready; tile[buf^1] drained
        if (kk + 32 < K) {
            copy16_g2l(Wb + (size_t)(nblk + col0) * K + kk + 32 + part0, &Wt[buf ^ 1][col0 * 32 + part0]);
            copy16_g2l(Wb + (size_t)(nblk + col1) * K + kk + 32 + part1, &Wt[buf ^ 1][col1 * 32 + part1]);
        }
        FragU aF;
        const unsigned short* ap = A + (size_t)am * K + kk + hsel;
        aF.q[0] = *(const u32x4*)(ap);
        aF.q[1] = *(const u32x4*)(ap + 16);
        if (kk + 64 < K) __builtin_prefetch(ap + 64, 0, 0);   // global_prefetch next A line
        #pragma unroll
        for (int s = 0; s < 4; ++s) {
            FragU bF;
            const unsigned short* bp = &Wt[buf][(nwave + s * 16 + (lane & 15)) * 32 + hsel];
            bF.q[0] = *(const u32x4*)(bp);
            bF.q[1] = *(const u32x4*)(bp + 16);
            acc[s] = __builtin_amdgcn_wmma_f32_16x16x32_bf16(
                false, aF.v, false, bF.v, (short)0, acc[s], false, false);
        }
    }
    #pragma unroll
    for (int s = 0; s < 4; ++s)
        #pragma unroll
        for (int r = 0; r < 8; ++r) {
            int m   = mbase + r + hsel;
            int col = nblk + nwave + s * 16 + (lane & 15);
            float val = acc[s][r] + bias[col];
            if (EPI == 0) {
                ((unsigned short*)outp)[(size_t)m * 256 + col] = f2bf(val * SCALE_Q);
            } else if (EPI == 1) {
                ((unsigned short*)outp)[(size_t)m * 512 + col] = f2bf(val);
            } else if (EPI == 2) {
                int g = m >> 6, n = m & 63;
                int b = g >> 6, w = g & 63;
                int is = (w >> 3) * 8 + (n >> 3), js = (w & 7) * 8 + (n & 7);
                int i = (is + 4) & 63, j = (js + 4) & 63;
                size_t dst = ((size_t)(b << 12) + (i << 6) + j) * 256 + col;
                ((float*)outp)[dst] = aux[dst] + val;
            } else if (EPI == 3) {
                float ge = 0.5f * val * (1.0f + erff(val * 0.70710678118654752f));
                ((unsigned short*)outp)[(size_t)m * 1024 + col] = f2bf(ge);
            } else {
                size_t dst = (size_t)m * 256 + col;
                ((float*)outp)[dst] = aux[dst] + val;
            }
        }
}

// ---------------- attention: one block per window, one wave per head ------------
// Phase 1 computes T = S^T = K @ Q^T so each softmax row (fixed query = lane) is
// 32 in-lane values + one SWAPX16 exchange with lane^16.
__global__ __launch_bounds__(256)
void attn_kernel(const unsigned short* __restrict__ qb,
                 const unsigned short* __restrict__ kvb,
                 const float* __restrict__ biasT,
                 unsigned short* __restrict__ aob) {
    __shared__ unsigned short P[8][64][64];            // 64 KB: P[query][key] per head
    int lane = threadIdx.x & 31, h = threadIdx.x >> 5;
    int hsel = (lane >> 4) << 3;
    int g = blockIdx.x;
    int w = g & 63, wrow = w >> 3, wcol = w & 7;
    const unsigned short* qh = qb  + (size_t)g * NTOK * 256 + h * DHEAD;
    const unsigned short* kh = kvb + (size_t)g * NTOK * 512 + h * DHEAD;
    const unsigned short* vh = kvb + (size_t)g * NTOK * 512 + 256 + h * DHEAD;
    const float* bh = biasT + (h << 12);

    // pack this lane's 32 key region-codes, 4 bits each (window-dependent, nt-invariant)
    unsigned ckPack[4];
    #pragma unroll
    for (int mt = 0; mt < 4; ++mt) {
        unsigned pk = 0;
        #pragma unroll
        for (int r = 0; r < 8; ++r) {
            int m = mt * 16 + hsel + r;                 // key token
            int ik = wrow * 8 + (m >> 3), jk = wcol * 8 + (m & 7);
            unsigned ck = 3u * (ik < 56 ? 0 : (ik < 60 ? 1 : 2)) + (jk < 56 ? 0 : (jk < 60 ? 1 : 2));
            pk |= ck << (4 * r);
        }
        ckPack[mt] = pk;
    }

    // ---- phase 1: per query tile nt, scores over all 64 keys, softmax, P->LDS ----
    for (int nt = 0; nt < 4; ++nt) {
        int qtok = nt * 16 + (lane & 15);                    // this lane's query token
        FragU bQ;                                            // B = Q^T : B[k=d][n=query]
        {
            const unsigned short* bp = qh + (size_t)qtok * 256 + hsel;
            bQ.q[0] = *(const u32x4*)(bp);
            bQ.q[1] = *(const u32x4*)(bp + 16);
        }
        v8f accT[4];
        #pragma unroll
        for (int mt = 0; mt < 4; ++mt) {
            FragU aK;                                        // A = K : A[m=key][k=d]
            const unsigned short* ap = kh + (size_t)(mt * 16 + (lane & 15)) * 512 + hsel;
            aK.q[0] = *(const u32x4*)(ap);
            aK.q[1] = *(const u32x4*)(ap + 16);
            v8f z = {};
            accT[mt] = __builtin_amdgcn_wmma_f32_16x16x32_bf16(
                false, aK.v, false, bQ.v, (short)0, z, false, false);
        }
        // bias (vectorized b128 loads of contiguous keys) + mask (packed codes)
        int iq = wrow * 8 + (qtok >> 3), jq = wcol * 8 + (qtok & 7);
        unsigned cq = 3u * (iq < 56 ? 0 : (iq < 60 ? 1 : 2)) + (jq < 56 ? 0 : (jq < 60 ? 1 : 2));
        #pragma unroll
        for (int mt = 0; mt < 4; ++mt) {
            const float* bp2 = bh + (qtok << 6) + mt * 16 + hsel;   // 8 consecutive keys
            f32x4 b0 = *(const f32x4*)(bp2);
            f32x4 b1 = *(const f32x4*)(bp2 + 4);
            float bv[8] = {b0.x, b0.y, b0.z, b0.w, b1.x, b1.y, b1.z, b1.w};
            #pragma unroll
            for (int r = 0; r < 8; ++r) {
                unsigned ck = (ckPack[mt] >> (4 * r)) & 15u;
                accT[mt][r] += bv[r] + (ck == cq ? 0.f : -100.f);
            }
        }
        // softmax over keys: in-lane + one xor-16 exchange
        float mx = -1e30f;
        #pragma unroll
        for (int mt = 0; mt < 4; ++mt)
            #pragma unroll
            for (int r = 0; r < 8; ++r) mx = fmaxf(mx, accT[mt][r]);
        mx = fmaxf(mx, swz_xor<16>(mx));
        float sum = 0.f;
        #pragma unroll
        for (int mt = 0; mt < 4; ++mt)
            #pragma unroll
            for (int r = 0; r < 8; ++r) { float pp = __expf(accT[mt][r] - mx); accT[mt][r] = pp; sum += pp; }
        sum += swz_xor<16>(sum);
        float inv = 1.0f / sum;
        // write P[query][key]: 8 keys contiguous -> one b128 store
        #pragma unroll
        for (int mt = 0; mt < 4; ++mt) {
            FragU pk;
            #pragma unroll
            for (int r = 0; r < 8; r += 2) {
                unsigned lo = f2bf(accT[mt][r] * inv);
                unsigned hi = f2bf(accT[mt][r + 1] * inv);
                pk.u[r >> 1] = lo | (hi << 16);
            }
            *(u32x4*)&P[h][qtok][mt * 16 + hsel] = pk.q[0];
        }
    }
    __syncthreads();

    // ---- phase 2: O = P @ V ----
    for (int mt = 0; mt < 4; ++mt) {
        v8f accO[2] = {};
        #pragma unroll
        for (int ks = 0; ks < 2; ++ks) {
            FragU aP;                                        // A = P rows (queries)
            const unsigned short* ap = &P[h][mt * 16 + (lane & 15)][ks * 32 + hsel];
            aP.q[0] = *(const u32x4*)(ap);
            aP.q[1] = *(const u32x4*)(ap + 16);
            #pragma unroll
            for (int t2 = 0; t2 < 2; ++t2) {
                FragU bV;                                    // B[k=token][n=d] gathers
                #pragma unroll
                for (int vv = 0; vv < 8; ++vv) {
                    int k = ks * 32 + ((vv >> 2) << 4) + hsel + ((vv & 3) << 1);
                    int n = t2 * 16 + (lane & 15);
                    unsigned lo = vh[(size_t)k * 512 + n];
                    unsigned hi = vh[(size_t)(k + 1) * 512 + n];
                    bV.u[vv] = lo | (hi << 16);
                }
                accO[t2] = __builtin_amdgcn_wmma_f32_16x16x32_bf16(
                    false, aP.v, false, bV.v, (short)0, accO[t2], false, false);
            }
        }
        #pragma unroll
        for (int t2 = 0; t2 < 2; ++t2)
            #pragma unroll
            for (int r = 0; r < 8; ++r) {
                int m = mt * 16 + r + hsel;                  // query token
                int d = t2 * 16 + (lane & 15);
                aob[((size_t)g * NTOK + m) * 256 + h * DHEAD + d] = f2bf(accO[t2][r]);
            }
    }
}

// ---------------- launcher ----------------
extern "C" void kernel_launch(void* const* d_in, const int* in_sizes, int n_in,
                              void* d_out, int out_size, void* d_ws, size_t ws_size,
                              hipStream_t stream) {
    const float* x0   = (const float*)d_in[0];
    const float* x1   = (const float*)d_in[1];
    const float* g1_0 = (const float*)d_in[2];
    const float* b1_0 = (const float*)d_in[3];
    const float* g1_1 = (const float*)d_in[4];
    const float* b1_1 = (const float*)d_in[5];
    const float* Wq   = (const float*)d_in[6];
    const float* bq   = (const float*)d_in[7];
    const float* Wkv  = (const float*)d_in[8];
    const float* bkv  = (const float*)d_in[9];
    const float* rpb  = (const float*)d_in[10];
    const float* Wp   = (const float*)d_in[11];
    const float* bp   = (const float*)d_in[12];
    const float* g2   = (const float*)d_in[13];
    const float* b2   = (const float*)d_in[14];
    const float* Wfc1 = (const float*)d_in[15];
    const float* bfc1 = (const float*)d_in[16];
    const float* Wfc2 = (const float*)d_in[17];
    const float* bfc2 = (const float*)d_in[18];

    char* ws = (char*)d_ws;
    const size_t MB = 1u << 20;
    unsigned short* x1w = (unsigned short*)(ws + 0);          // 32 MB
    unsigned short* x0w = (unsigned short*)(ws + 32 * MB);    // 32 MB
    unsigned short* qbf = (unsigned short*)(ws + 64 * MB);    // 32 MB
    unsigned short* kvb = (unsigned short*)(ws + 96 * MB);    // 64 MB
    unsigned short* hbf = (unsigned short*)(ws + 0);          // 128 MB (aliases dead bufs)
    unsigned short* aob = (unsigned short*)(ws + 160 * MB);   // 32 MB
    float*          xres= (float*)         (ws + 192 * MB);   // 64 MB
    unsigned short* xln = (unsigned short*)(ws + 256 * MB);   // 32 MB
    unsigned short* wqB = (unsigned short*)(ws + 288 * MB);
    unsigned short* wkvB= wqB  + 65536;
    unsigned short* wpB = wkvB + 131072;
    unsigned short* wf1B= wpB  + 65536;
    unsigned short* wf2B= wf1B + 262144;
    float*          biasT=(float*)(ws + 290 * MB);            // 128 KB

    cast_f32_bf16<<<(65536 + 255) / 256, 256, 0, stream>>>(Wq,   wqB,  65536);
    cast_f32_bf16<<<(131072 + 255) / 256, 256, 0, stream>>>(Wkv,  wkvB, 131072);
    cast_f32_bf16<<<(65536 + 255) / 256, 256, 0, stream>>>(Wp,   wpB,  65536);
    cast_f32_bf16<<<(262144 + 255) / 256, 256, 0, stream>>>(Wfc1, wf1B, 262144);
    cast_f32_bf16<<<(262144 + 255) / 256, 256, 0, stream>>>(Wfc2, wf2B, 262144);
    bias_prep<<<128, 256, 0, stream>>>(rpb, biasT);

    ln_shift_window<<<NROWS, 256, 0, stream>>>(x0, x1, g1_0, b1_0, g1_1, b1_1, x0w, x1w);

    gemm_bf16_ws<256, 0><<<dim3(1024, 2), 256, 0, stream>>>(x1w, wqB,  bq,   nullptr, qbf);
    gemm_bf16_ws<256, 1><<<dim3(1024, 4), 256, 0, stream>>>(x0w, wkvB, bkv,  nullptr, kvb);

    attn_kernel<<<1024, 256, 0, stream>>>(qbf, kvb, biasT, aob);

    gemm_bf16_ws<256, 2><<<dim3(1024, 2), 256, 0, stream>>>(aob, wpB,  bp,   x1,   xres);

    ln2_kernel<<<NROWS, 256, 0, stream>>>(xres, g2, b2, xln);

    gemm_bf16_ws<256, 3><<<dim3(1024, 8), 256, 0, stream>>>(xln, wf1B, bfc1, nullptr, hbf);
    gemm_bf16_ws<1024, 4><<<dim3(1024, 2), 256, 0, stream>>>(hbf, wf2B, bfc2, xres, (float*)d_out);
}
